// GausSplatingHead_31275951850106
// MI455X (gfx1250) — matrix operand order repaired and checked by reading, not compiled, not run on hardware
//
#include <hip/hip_runtime.h>
#include <hip/hip_bf16.h>

#define N_GAUSS 3200
#define FEAT_DIM 17
#define RW 88
#define RH 48
#define NPIX (RW*RH)          // 4224
#define GT_H 192
#define GT_W 352
#define KT 32                 // gaussians per k-tile
#define NT (N_GAUSS/KT)       // 100 tiles
#define LFS 40                // LDS feats row stride in halfs (80B, bank-friendly, 16B-aligned)
#define NBLK 33               // 33 blocks * 128 px = 4224

typedef __attribute__((ext_vector_type(16))) _Float16 v16h;
typedef __attribute__((ext_vector_type(8)))  _Float16 v8h;
typedef __attribute__((ext_vector_type(8)))  float    v8f;

// ---- workspace layout (bytes) ----
#define OFF_PART 0                       // 2*33*2 f32 block partials (528B) -> pad 1024
#define OFF_Z    1024                    // z[2][3200] f32             = 25600
#define OFF_UP   (OFF_Z + 25600)         // unsorted params [2][3200][8] f32 = 204800
#define OFF_SP   (OFF_UP + 204800)       // sorted params  [2][3200][8] f32 = 204800
#define OFF_SF   (OFF_SP + 204800)       // sorted featsT  [2][32][3200] f16 = 409600

// ------------------------------------------------------------------
// Phase 1: per-(camera, gaussian) projection -> u,v, conic, z, op*valid
// ------------------------------------------------------------------
__global__ void proj_kernel(const float* __restrict__ xyz, const float* __restrict__ scales,
                            const float* __restrict__ rots, const float* __restrict__ vms,
                            const float* __restrict__ intr, const float* __restrict__ opac,
                            float* __restrict__ zbuf, float* __restrict__ up)
{
    int g = blockIdx.x * blockDim.x + threadIdx.x;
    int cam = blockIdx.y;
    if (g >= N_GAUSS) return;

    // quaternion -> rotation
    float qw = rots[g*4+0], qx = rots[g*4+1], qy = rots[g*4+2], qz = rots[g*4+3];
    float qn = rsqrtf(qw*qw + qx*qx + qy*qy + qz*qz);
    qw *= qn; qx *= qn; qy *= qn; qz *= qn;
    float R[3][3] = {
        {1.f-2.f*(qy*qy+qz*qz), 2.f*(qx*qy-qw*qz),     2.f*(qx*qz+qw*qy)},
        {2.f*(qx*qy+qw*qz),     1.f-2.f*(qx*qx+qz*qz), 2.f*(qy*qz-qw*qx)},
        {2.f*(qx*qz-qw*qy),     2.f*(qy*qz+qw*qx),     1.f-2.f*(qx*qx+qy*qy)}};
    float ss[3];
    #pragma unroll
    for (int j = 0; j < 3; j++) { float s = __expf(scales[g*3+j]); ss[j] = s*s; }

    // cov3d = R diag(ss) R^T
    float cov3[3][3];
    #pragma unroll
    for (int i = 0; i < 3; i++)
        #pragma unroll
        for (int k = 0; k < 3; k++)
            cov3[i][k] = R[i][0]*ss[0]*R[k][0] + R[i][1]*ss[1]*R[k][1] + R[i][2]*ss[2]*R[k][2];

    const float* vm = vms + cam*16;
    float Rw[3][3], t[3];
    #pragma unroll
    for (int i = 0; i < 3; i++) {
        Rw[i][0] = vm[i*4+0]; Rw[i][1] = vm[i*4+1]; Rw[i][2] = vm[i*4+2];
        t[i] = vm[i*4+3];
    }
    float X = xyz[g*3+0], Y = xyz[g*3+1], Z = xyz[g*3+2];
    float p[3];
    #pragma unroll
    for (int i = 0; i < 3; i++) p[i] = Rw[i][0]*X + Rw[i][1]*Y + Rw[i][2]*Z + t[i];

    // covc = Rw cov3 Rw^T
    float tm[3][3], covc[3][3];
    #pragma unroll
    for (int i = 0; i < 3; i++)
        #pragma unroll
        for (int k = 0; k < 3; k++)
            tm[i][k] = Rw[i][0]*cov3[0][k] + Rw[i][1]*cov3[1][k] + Rw[i][2]*cov3[2][k];
    #pragma unroll
    for (int i = 0; i < 3; i++)
        #pragma unroll
        for (int l = 0; l < 3; l++)
            covc[i][l] = tm[i][0]*Rw[l][0] + tm[i][1]*Rw[l][1] + tm[i][2]*Rw[l][2];

    float fx = intr[cam*4+0], fy = intr[cam*4+1], cx = intr[cam*4+2], cy = intr[cam*4+3];
    float z  = p[2];
    float zc = fmaxf(z, 0.2f);
    float u  = fx*p[0]/zc + cx;
    float v  = fy*p[1]/zc + cy;

    float j00 = fx/zc, j02 = -fx*p[0]/(zc*zc);
    float j11 = fy/zc, j12 = -fy*p[1]/(zc*zc);
    // cov2 = J covc J^T ; J rows (j00,0,j02), (0,j11,j12)
    float r0c0 = j00*covc[0][0] + j02*covc[2][0];
    float r0c1 = j00*covc[0][1] + j02*covc[2][1];
    float r0c2 = j00*covc[0][2] + j02*covc[2][2];
    float r1c0 = j11*covc[1][0] + j12*covc[2][0];
    float r1c1 = j11*covc[1][1] + j12*covc[2][1];
    float r1c2 = j11*covc[1][2] + j12*covc[2][2];
    float c00 = r0c0*j00 + r0c2*j02;
    float c01 = r0c1*j11 + r0c2*j12;
    float c11 = r1c1*j11 + r1c2*j12;

    float a = c00 + 0.3f, b = c01, c = c11 + 0.3f;
    float det = fmaxf(a*c - b*b, 1e-8f);
    float Ai =  c/det, Bi = -b/det, Ci = a/det;
    float ov = (z > 0.2f) ? opac[g] : 0.f;

    zbuf[cam*N_GAUSS + g] = z;
    float* o = up + (size_t)(cam*N_GAUSS + g)*8;
    o[0] = u; o[1] = v; o[2] = Ai; o[3] = Bi; o[4] = Ci; o[5] = ov; o[6] = 0.f; o[7] = 0.f;
}

// ------------------------------------------------------------------
// Phase 2: stable rank sort by z (count), scatter params + f16 featsT
// ------------------------------------------------------------------
__global__ void sort_scatter_kernel(const float* __restrict__ zbuf, const float* __restrict__ up,
                                    const float* __restrict__ feats,
                                    float* __restrict__ sp, _Float16* __restrict__ sf)
{
    __shared__ float zs[N_GAUSS];
    int cam = blockIdx.y;
    for (int i = threadIdx.x; i < N_GAUSS; i += blockDim.x) zs[i] = zbuf[cam*N_GAUSS + i];
    __syncthreads();

    int g = blockIdx.x * blockDim.x + threadIdx.x;
    if (g >= N_GAUSS) return;
    float zg = zs[g];
    int rank = 0;
    for (int j = 0; j < N_GAUSS; j++) {
        float zj = zs[j];
        rank += (zj < zg) || (zj == zg && j < g);   // stable ascending argsort position
    }
    const float* src = up + (size_t)(cam*N_GAUSS + g)*8;
    float*       dst = sp + (size_t)(cam*N_GAUSS + rank)*8;
    #pragma unroll
    for (int k = 0; k < 8; k++) dst[k] = src[k];
    _Float16* fcam = sf + (size_t)cam*32*N_GAUSS;
    #pragma unroll
    for (int ch = 0; ch < 32; ch++)
        fcam[ch*N_GAUSS + rank] = (ch < FEAT_DIM) ? (_Float16)feats[g*FEAT_DIM + ch] : (_Float16)0.f;
}

// ------------------------------------------------------------------
// Phase 3: tiled alpha-composite via WMMA + masked weighted CE
// ------------------------------------------------------------------
__global__ __launch_bounds__(256, 1) void render_ce_kernel(
    const float* __restrict__ sp, const _Float16* __restrict__ sf,
    const int* __restrict__ gt_sem, const int* __restrict__ sem_mask,
    const float* __restrict__ cw, float* __restrict__ part)
{
    __shared__ float sP[KT*8];                       // 32 gaussians x 8 params
    __shared__ __align__(16) _Float16 sF[32*LFS];    // 32 channels x 32 gaussians (padded)
    __shared__ float sImg[8*16*33];                  // 8 waves x 16 px x 32 ch (pad 33)
    __shared__ float sSum[16];
    __shared__ float sCW[FEAT_DIM];

    const int tid  = threadIdx.x;
    const int cam  = blockIdx.y;
    const int wave = tid >> 5;
    const int lane = tid & 31;
    if (tid < FEAT_DIM) sCW[tid] = cw[tid];

    const int  pixBase = blockIdx.x*128 + wave*16;
    const bool high = lane >= 16;
    const int  m    = lane & 15;        // A-row (pixel) / B,C column (channel)
    const int  pix  = pixBase + m;
    const float px  = (float)(pix % RW);
    const float py  = (float)(pix / RW);
    const int  kb0  = high ? 8 : 0;     // A-matrix K sub-block for this half-wave
    const int  kbB  = high ? 8 : 0;     // B-matrix K sub-block

    const float*    gp = sp + (size_t)cam*N_GAUSS*8;
    const _Float16* gf = sf + (size_t)cam*32*N_GAUSS;

    v8f acc0 = {}; v8f acc1 = {};
    float Tc = 1.0f;                    // per-pixel transmittance carry

    for (int t = 0; t < NT; t++) {
        __syncthreads();
        // stage params: 256 contiguous floats
        sP[tid] = gp[t*KT*8 + tid];
        // stage features transposed: thread -> (channel, 4 gaussians)
        {
            int ch = tid >> 3, pr = tid & 7;
            uint2 d = *(const uint2*)(gf + ch*N_GAUSS + t*KT + pr*4);
            *(uint2*)(&sF[ch*LFS + pr*4]) = d;
        }
        if (t + 1 < NT)                 // gfx1250 global_prefetch of next tile
            __builtin_prefetch(gp + (t+1)*KT*8 + tid, 0, 1);
        __syncthreads();

        // per-lane alphas for 16 (pixel,k) pairs in A-matrix element order,
        // with in-lane prefix products of (1-alpha) per group of 8
        float w[16];
        float run = 1.0f, q0 = 1.0f, q1;
        #pragma unroll
        for (int i = 0; i < 16; i++) {
            if (i == 8) { q0 = run; run = 1.0f; }
            int koff = (i < 8) ? (kb0 + i) : (kb0 + 16 + (i - 8));
            const float* pp = &sP[koff*8];           // LDS broadcast across 16 lanes
            float dx = px - pp[0];
            float dy = py - pp[1];
            float pw = -0.5f*(pp[2]*dx*dx + pp[4]*dy*dy) - pp[3]*dx*dy;
            pw = fminf(pw, 0.0f);
            float al = fminf(pp[5]*__expf(pw), 0.99f);
            w[i] = al * run;
            run *= (1.0f - al);
        }
        q1 = run;
        // lane-pair exchange: pixel's K-order is loG0, hiG0, loG1, hiG1
        float q0o = __shfl_xor(q0, 16);
        float q1o = __shfl_xor(q1, 16);
        float m0 = high ? Tc*q0o           : Tc;
        float m1 = high ? Tc*q0o*q0*q1o    : Tc*q0*q0o;
        Tc *= q0*q1*q0o*q1o;

        v16h Aop;
        #pragma unroll
        for (int i = 0; i < 8;  i++) Aop[i] = (_Float16)(w[i]*m0);
        #pragma unroll
        for (int i = 8; i < 16; i++) Aop[i] = (_Float16)(w[i]*m1);

        // B operands: channels [0..15] and [16..31] (17..31 are zero pads)
        v8h b0a = *(const v8h*)&sF[m*LFS + kbB];
        v8h b0b = *(const v8h*)&sF[m*LFS + kbB + 16];
        v8h b1a = *(const v8h*)&sF[(16+m)*LFS + kbB];
        v8h b1b = *(const v8h*)&sF[(16+m)*LFS + kbB + 16];
        v16h B0 = __builtin_shufflevector(b0a, b0b, 0,1,2,3,4,5,6,7,8,9,10,11,12,13,14,15);
        v16h B1 = __builtin_shufflevector(b1a, b1b, 0,1,2,3,4,5,6,7,8,9,10,11,12,13,14,15);

        acc0 = __builtin_amdgcn_wmma_f32_16x16x32_f16(false, Aop, false, B0, (short)0, acc0, false, false);
        acc1 = __builtin_amdgcn_wmma_f32_16x16x32_f16(false, Aop, false, B1, (short)0, acc1, false, false);
    }

    // spill D-matrix (pixels x channels) to LDS per C/D layout
    float* wimg = &sImg[wave*16*33];
    {
        int prow = high ? 8 : 0;
        #pragma unroll
        for (int r = 0; r < 8; r++) {
            wimg[(prow + r)*33 + m]      = acc0[r];
            wimg[(prow + r)*33 + 16 + m] = acc1[r];
        }
    }
    __syncthreads();

    // masked weighted CE: lanes 0..15 each own one pixel
    float num = 0.f, den = 0.f;
    if (lane < 16) {
        int p2 = pixBase + lane;
        int x = p2 % RW, y = p2 / RW;
        int gi = cam*GT_H*GT_W + (y*4)*GT_W + (x*4);   // nearest resize: stride-4 sample
        int gtl = gt_sem[gi];
        float msk = (float)sem_mask[gi];
        float wi = sCW[gtl] * msk;
        const float* lg = &wimg[lane*33];
        float mx = lg[0];
        #pragma unroll
        for (int ch = 1; ch < FEAT_DIM; ch++) mx = fmaxf(mx, lg[ch]);
        float se = 0.f;
        #pragma unroll
        for (int ch = 0; ch < FEAT_DIM; ch++) se += __expf(lg[ch] - mx);
        float lse = mx + __logf(se);
        float nll = lse - lg[gtl];
        num = wi * nll; den = wi;
    }
    #pragma unroll
    for (int off = 16; off >= 1; off >>= 1) {          // fixed-tree wave reduction
        num += __shfl_xor(num, off);
        den += __shfl_xor(den, off);
    }
    if (lane == 0) { sSum[wave*2] = num; sSum[wave*2+1] = den; }
    __syncthreads();
    if (tid == 0) {                                    // fixed-order block reduction
        float n = 0.f, d = 0.f;
        #pragma unroll
        for (int wv = 0; wv < 8; wv++) { n += sSum[wv*2]; d += sSum[wv*2+1]; }
        part[(cam*NBLK + blockIdx.x)*2 + 0] = n;
        part[(cam*NBLK + blockIdx.x)*2 + 1] = d;
    }
}

// ------------------------------------------------------------------
// Phase 4: deterministic final reduction -> scalar loss
// ------------------------------------------------------------------
__global__ void finalize_kernel(const float* __restrict__ part, float* __restrict__ out)
{
    if (threadIdx.x == 0 && blockIdx.x == 0) {
        float loss = 0.f;
        for (int cam = 0; cam < 2; cam++) {
            float n = 0.f, d = 0.f;
            for (int b = 0; b < NBLK; b++) {
                n += part[(cam*NBLK + b)*2 + 0];
                d += part[(cam*NBLK + b)*2 + 1];
            }
            loss += n / fmaxf(d, 1e-8f);
        }
        out[0] = loss * 0.5f;   // / n_cam, B == 1
    }
}

extern "C" void kernel_launch(void* const* d_in, const int* in_sizes, int n_in,
                              void* d_out, int out_size, void* d_ws, size_t ws_size,
                              hipStream_t stream)
{
    const float* feats  = (const float*)d_in[0];  // (1,3200,17)
    const float* opac   = (const float*)d_in[1];  // (1,3200,1)
    const float* xyz    = (const float*)d_in[2];  // (3200,3)
    const float* scales = (const float*)d_in[3];  // (3200,3)
    const float* rots   = (const float*)d_in[4];  // (3200,4)
    const float* vms    = (const float*)d_in[5];  // (2,4,4)
    const float* intr   = (const float*)d_in[6];  // (2,4)
    const float* cw     = (const float*)d_in[7];  // (17,)
    const int*   gt     = (const int*)d_in[8];    // (2,192,352)
    const int*   mask   = (const int*)d_in[9];    // (2,192,352)

    char* ws = (char*)d_ws;
    float*    partb = (float*)(ws + OFF_PART);
    float*    zbuf  = (float*)(ws + OFF_Z);
    float*    up    = (float*)(ws + OFF_UP);
    float*    spar  = (float*)(ws + OFF_SP);
    _Float16* sfeat = (_Float16*)(ws + OFF_SF);

    proj_kernel<<<dim3(13, 2), 256, 0, stream>>>(xyz, scales, rots, vms, intr, opac, zbuf, up);
    sort_scatter_kernel<<<dim3(13, 2), 256, 0, stream>>>(zbuf, up, feats, spar, sfeat);
    render_ce_kernel<<<dim3(NBLK, 2), 256, 0, stream>>>(spar, sfeat, gt, mask, cw, partb);
    finalize_kernel<<<1, 1, 0, stream>>>(partb, (float*)d_out);
}
